// smallGPT_8117488189849
// MI455X (gfx1250) — matrix operand order repaired and checked by reading, not compiled, not run on hardware
//
#include <hip/hip_runtime.h>
#include <hip/hip_bf16.h>
#include <math.h>

// ---------------------------------------------------------------------------
// Types for CDNA5 WMMA (wave32): A/B = 16 bf16 per lane, C/D = 8 f32 per lane
// ---------------------------------------------------------------------------
typedef __attribute__((ext_vector_type(16))) __bf16       v16bf;
typedef __attribute__((ext_vector_type(8)))  float        v8f;
typedef __attribute__((ext_vector_type(4)))  unsigned int u32x4;

union FragA { v16bf v; unsigned int u[8]; __bf16 e[16]; };
union AccF  { v8f   v; float f[8]; };
union B8    { u32x4 u4; __bf16 e[8]; };

static __device__ inline v8f wmma_bf16(v16bf a, v16bf b, v8f c) {
    // (neg_a, A, neg_b, B, c_mod, C, reuse_a, reuse_b)
    return __builtin_amdgcn_wmma_f32_16x16x32_bf16(false, a, false, b,
                                                   (short)0, c, false, false);
}

// Async global->LDS copy of 16 bytes per lane (tracked by ASYNCcnt).
// ldsoff = low 32 bits of generic LDS pointer (aperture: addr[31:0] = offset).
static __device__ inline void async_ld_b128(unsigned ldsoff, const void* gptr) {
    asm volatile("global_load_async_to_lds_b128 %0, %1, off"
                 :: "v"(ldsoff), "v"((unsigned long long)(size_t)gptr)
                 : "memory");
}
static __device__ inline void wait_async0() {
    asm volatile("s_wait_asynccnt 0x0" ::: "memory");
}

// ---------------------------------------------------------------------------
// fp32 -> bf16 convert (weights)
// ---------------------------------------------------------------------------
__global__ void cvt_kernel(const float* __restrict__ in, __bf16* __restrict__ out, int n) {
    int i = blockIdx.x * blockDim.x + threadIdx.x;
    int stride = gridDim.x * blockDim.x;
    for (; i < n; i += stride) out[i] = (__bf16)in[i];
}

// ---------------------------------------------------------------------------
// Embedding: x[b,t,:] = tok_emb[inp_x[b,t],:] + pos_emb[t,:]
// ---------------------------------------------------------------------------
__global__ void embed_kernel(const int* __restrict__ ix, const float* __restrict__ tok,
                             const float* __restrict__ pos, float* __restrict__ x,
                             int T, int Cdim) {
    int gid = blockIdx.x * blockDim.x + threadIdx.x;
    int row = gid / Cdim, c = gid % Cdim;
    int tpos = row % T;
    x[gid] = tok[(size_t)ix[row] * Cdim + c] + pos[(size_t)tpos * Cdim + c];
}

// ---------------------------------------------------------------------------
// LayerNorm over C=512, one row per 128-thread block, bf16 output
// ---------------------------------------------------------------------------
__global__ __launch_bounds__(128) void ln_kernel(const float* __restrict__ x,
                                                 const float* __restrict__ g,
                                                 const float* __restrict__ b,
                                                 __bf16* __restrict__ out, int Cdim) {
    __shared__ float ssum[128], ssq[128];
    const int row = blockIdx.x, t = threadIdx.x;
    const float4 vv = ((const float4*)(x + (size_t)row * Cdim))[t];
    ssum[t] = vv.x + vv.y + vv.z + vv.w;
    ssq[t]  = vv.x*vv.x + vv.y*vv.y + vv.z*vv.z + vv.w*vv.w;
    __syncthreads();
    for (int s = 64; s > 0; s >>= 1) {
        if (t < s) { ssum[t] += ssum[t+s]; ssq[t] += ssq[t+s]; }
        __syncthreads();
    }
    const float mean = ssum[0] / (float)Cdim;
    const float var  = ssq[0] / (float)Cdim - mean * mean;
    const float rs   = rsqrtf(var + 1e-5f);
    #pragma unroll
    for (int j = 0; j < 4; j++) {
        int c = t * 4 + j;
        float e = ((&vv.x)[j] - mean) * rs * g[c] + b[c];
        out[(size_t)row * Cdim + c] = (__bf16)e;
    }
}

// ---------------------------------------------------------------------------
// Tiled WMMA GEMM, double-buffered async-to-LDS A staging, pipelined B.
// out[M,N] = A[M,K](bf16) @ B[K,N](bf16) (+bias)(+GELU)(+res)
// Block = 256 threads (8 waves), tile 128x128, BK=32.
// Waves arranged 4(M) x 2(N); each wave computes 32x64 = 2x4 WMMA tiles.
// ---------------------------------------------------------------------------
template<bool BIAS, bool GELU_, bool RES, bool OUTBF>
__global__ __launch_bounds__(256) void gemm_bf16_kernel(
        const __bf16* __restrict__ A, const __bf16* __restrict__ Bm,
        const float* __restrict__ bias, const float* __restrict__ res,
        void* __restrict__ outp, int M, int N, int K) {
    __shared__ __align__(16) __bf16 sA[2][128 * 40];   // row-major [128][32] pad 8
    __shared__ __align__(16) __bf16 sBT[2][128 * 40];  // transposed [N=128][K=32] pad 8
    const int t    = threadIdx.x;
    const int lane = t & 31, wvid = t >> 5;
    const int wm   = wvid >> 1, wn = wvid & 1;         // 4 x 2 wave grid
    const int half = lane >> 4, l16 = lane & 15;
    const int m0   = blockIdx.x * 128, n0 = blockIdx.y * 128;

    const int rA  = t >> 2,         segA  = t & 3;     // A stage: 2 x async b128
    const int rA2 = (t + 256) >> 2, segA2 = (t + 256) & 3;

    auto issueA = [&](int buf, int k0) {
        async_ld_b128((unsigned)(size_t)&sA[buf][rA * 40 + segA * 8],
                      A + (size_t)(m0 + rA) * K + k0 + segA * 8);
        async_ld_b128((unsigned)(size_t)&sA[buf][rA2 * 40 + segA2 * 8],
                      A + (size_t)(m0 + rA2) * K + k0 + segA2 * 8);
    };
    // B stage: 2 chunks of 8 elems per thread (tile 32x128)
    auto loadB = [&](int k0, B8 bv[2]) {
        #pragma unroll
        for (int i = 0; i < 2; i++) {
            int c = t + i * 256;
            int kr = c >> 4, nseg = c & 15;
            bv[i].u4 = *(const u32x4*)(Bm + (size_t)(k0 + kr) * N + n0 + nseg * 8);
        }
    };
    auto storeB = [&](int buf, B8 bv[2]) {
        #pragma unroll
        for (int i = 0; i < 2; i++) {
            int c = t + i * 256;
            int kr = c >> 4, nseg = c & 15;
            #pragma unroll
            for (int e = 0; e < 8; e++)
                sBT[buf][(nseg * 8 + e) * 40 + kr] = bv[i].e[e];
        }
    };

    AccF acc[2][4];
    #pragma unroll
    for (int a = 0; a < 2; a++)
        #pragma unroll
        for (int bb = 0; bb < 4; bb++)
            #pragma unroll
            for (int i = 0; i < 8; i++) acc[a][bb].f[i] = 0.f;

    const int nk = K >> 5;
    issueA(0, 0);
    B8 bv[2];
    loadB(0, bv);

    for (int ik = 0; ik < nk; ik++) {
        const int cur = ik & 1;
        wait_async0();                 // this wave's async A tile landed in LDS
        storeB(cur, bv);               // transpose-store B tile
        __syncthreads();               // all waves: tiles ready

        B8 bnext[2] = { bv[0], bv[1] };
        if (ik + 1 < nk) {             // prefetch next tiles during compute
            issueA(cur ^ 1, (ik + 1) * 32);
            loadB((ik + 1) * 32, bnext);
        }

        FragA af[2], bfg[4];
        #pragma unroll
        for (int sm = 0; sm < 2; sm++) {
            const unsigned int* bp =
                (const unsigned int*)(&sA[cur][(wm * 32 + sm * 16 + l16) * 40]);
            #pragma unroll
            for (int p = 0; p < 8; p++)
                af[sm].u[p] = bp[half * 8 + p];
        }
        #pragma unroll
        for (int sn = 0; sn < 4; sn++) {
            const unsigned int* bp =
                (const unsigned int*)(&sBT[cur][(wn * 64 + sn * 16 + l16) * 40]);
            #pragma unroll
            for (int p = 0; p < 8; p++)
                bfg[sn].u[p] = bp[half * 8 + p];
        }
        #pragma unroll
        for (int sn = 0; sn < 4; sn++)        // 8 independent accumulators
            #pragma unroll
            for (int sm = 0; sm < 2; sm++)
                acc[sm][sn].v = wmma_bf16(af[sm].v, bfg[sn].v, acc[sm][sn].v);

        bv[0] = bnext[0]; bv[1] = bnext[1];
    }

    // ---- epilogue: C layout = VGPR i -> row i + half*8, col = l16 ----
    #pragma unroll
    for (int sm = 0; sm < 2; sm++)
        #pragma unroll
        for (int sn = 0; sn < 4; sn++)
            #pragma unroll
            for (int i = 0; i < 8; i++) {
                int m = m0 + wm * 32 + sm * 16 + i + half * 8;
                int n = n0 + wn * 64 + sn * 16 + l16;
                float vout = acc[sm][sn].f[i];
                if constexpr (BIAS) vout += bias[n];
                if constexpr (GELU_) vout = 0.5f * vout * (1.f + erff(vout * 0.70710678118f));
                size_t idx = (size_t)m * N + n;
                if constexpr (RES) vout += res[idx];
                if constexpr (OUTBF) ((__bf16*)outp)[idx] = (__bf16)vout;
                else                 ((float*)outp)[idx]  = vout;
            }
}

// ---------------------------------------------------------------------------
// Flash attention: one wave per 16-query tile per (b, head). Online softmax,
// QK^T and P@V both via WMMA. hd = 64, key blocks of 32.
// ---------------------------------------------------------------------------
__global__ __launch_bounds__(32) void attn_kernel(
        const __bf16* __restrict__ q, const __bf16* __restrict__ k,
        const __bf16* __restrict__ v, __bf16* __restrict__ o,
        int T, int Cdim, float scale) {
    __shared__ __align__(16) __bf16 sp[16 * 32];   // P tile reshape buffer
    const int lane = threadIdx.x & 31;
    const int half = lane >> 4, l16 = lane & 15;
    const int q0 = blockIdx.x * 16;
    const int hh = blockIdx.y;
    const int b  = blockIdx.z;

    // Q fragments (16x64 = two K-chunks of 32), A layout, packed dword loads
    FragA qa[2];
    {
        const __bf16* qr = q + ((size_t)b * T + (q0 + l16)) * Cdim + hh * 64;
        #pragma unroll
        for (int kc = 0; kc < 2; kc++) {
            const unsigned int* up = (const unsigned int*)(qr + kc * 32);
            #pragma unroll
            for (int p = 0; p < 8; p++)
                qa[kc].u[p] = up[half * 8 + p];
        }
    }

    float mst[8], lst[8];
    AccF oacc[4];
    #pragma unroll
    for (int i = 0; i < 8; i++) { mst[i] = -1e30f; lst[i] = 0.f; }
    #pragma unroll
    for (int d = 0; d < 4; d++)
        #pragma unroll
        for (int i = 0; i < 8; i++) oacc[d].f[i] = 0.f;

    for (int kb = 0; kb <= q0 + 15; kb += 32) {
        // ---- load all 4 K fragments, then interleave WMMAs (fewer hazards) ----
        FragA kf[2][2];   // [nt][kc]
        #pragma unroll
        for (int nt = 0; nt < 2; nt++) {
            int key = kb + nt * 16 + l16; if (key >= T) key = T - 1;
            const __bf16* kr = k + ((size_t)b * T + key) * Cdim + hh * 64;
            #pragma unroll
            for (int kc = 0; kc < 2; kc++) {
                const unsigned int* up = (const unsigned int*)(kr + kc * 32);
                #pragma unroll
                for (int p = 0; p < 8; p++) kf[nt][kc].u[p] = up[half * 8 + p];
            }
        }
        AccF s[2];
        #pragma unroll
        for (int nt = 0; nt < 2; nt++)
            #pragma unroll
            for (int i = 0; i < 8; i++) s[nt].f[i] = 0.f;
        s[0].v = wmma_bf16(qa[0].v, kf[0][0].v, s[0].v);
        s[1].v = wmma_bf16(qa[0].v, kf[1][0].v, s[1].v);
        s[0].v = wmma_bf16(qa[1].v, kf[0][1].v, s[0].v);
        s[1].v = wmma_bf16(qa[1].v, kf[1][1].v, s[1].v);

        // ---- scale, causal mask, online softmax ----
        float p0[8], p1[8], al[8];
        #pragma unroll
        for (int i = 0; i < 8; i++) {
            int qr_ = q0 + i + half * 8;
            int k0_ = kb + l16, k1_ = kb + 16 + l16;
            float s0 = (k0_ <= qr_) ? s[0].f[i] * scale : -1e30f;
            float s1 = (k1_ <= qr_) ? s[1].f[i] * scale : -1e30f;
            float rm = fmaxf(s0, s1);
            #pragma unroll
            for (int mlt = 1; mlt < 16; mlt <<= 1)
                rm = fmaxf(rm, __shfl_xor(rm, mlt, 32));
            float mnew = fmaxf(mst[i], rm);
            float a  = __expf(mst[i] - mnew);
            float e0 = __expf(s0 - mnew), e1 = __expf(s1 - mnew);
            float rs = e0 + e1;
            #pragma unroll
            for (int mlt = 1; mlt < 16; mlt <<= 1)
                rs += __shfl_xor(rs, mlt, 32);
            lst[i] = lst[i] * a + rs;
            mst[i] = mnew;
            al[i] = a; p0[i] = e0; p1[i] = e1;
        }
        #pragma unroll
        for (int d = 0; d < 4; d++)
            #pragma unroll
            for (int i = 0; i < 8; i++) oacc[d].f[i] *= al[i];

        // ---- reshape P (C layout) -> A layout through LDS ----
        #pragma unroll
        for (int i = 0; i < 8; i++) {
            sp[(i + half * 8) * 32 + l16]      = (__bf16)p0[i];
            sp[(i + half * 8) * 32 + 16 + l16] = (__bf16)p1[i];
        }
        asm volatile("s_wait_dscnt 0" ::: "memory");  // single wave: order st->ld
        FragA pf;
        {
            const unsigned int* up = (const unsigned int*)sp + l16 * 16;
            #pragma unroll
            for (int p = 0; p < 8; p++)
                pf.u[p] = up[half * 8 + p];
        }

        // ---- O += P @ V (four 16-wide d tiles, independent accumulators) ----
        #pragma unroll
        for (int dt = 0; dt < 4; dt++) {
            FragA vf;  // B layout: key = kb + half*16 + j, col = dt*16 + l16
            #pragma unroll
            for (int j = 0; j < 16; j++) {
                int key = kb + half * 16 + j; if (key >= T) key = T - 1;
                vf.e[j] = v[((size_t)b * T + key) * Cdim + hh * 64 + dt * 16 + l16];
            }
            oacc[dt].v = wmma_bf16(pf.v, vf.v, oacc[dt].v);
        }
    }

    // ---- normalize and write o (bf16) ----
    #pragma unroll
    for (int dt = 0; dt < 4; dt++)
        #pragma unroll
        for (int i = 0; i < 8; i++) {
            int m = q0 + i + half * 8;
            float ov = oacc[dt].f[i] / lst[i];
            o[((size_t)b * T + m) * Cdim + hh * 64 + dt * 16 + l16] = (__bf16)ov;
        }
}

// ---------------------------------------------------------------------------
// Orchestration
// ---------------------------------------------------------------------------
extern "C" void kernel_launch(void* const* d_in, const int* in_sizes, int n_in,
                              void* d_out, int out_size, void* d_ws, size_t ws_size,
                              hipStream_t stream) {
    constexpr int Bb = 2, T = 2048, C = 512, H = 8, L = 4, FF = 2048, V = 32000;
    constexpr int BT = Bb * T;

    const int*   inp_x   = (const int*)d_in[0];
    const float* tok_emb = (const float*)d_in[2];
    const float* pos_emb = (const float*)d_in[3];
    const float* ln1_g   = (const float*)d_in[4];
    const float* ln1_b   = (const float*)d_in[5];
    const float* Wq      = (const float*)d_in[6];
    const float* Wk      = (const float*)d_in[7];
    const float* Wv      = (const float*)d_in[8];
    const float* Wproj   = (const float*)d_in[9];
    const float* bproj   = (const float*)d_in[10];
    const float* ln2_g   = (const float*)d_in[11];
    const float* ln2_b   = (const float*)d_in[12];
    const float* Wff1    = (const float*)d_in[13];
    const float* bff1    = (const float*)d_in[14];
    const float* Wff2    = (const float*)d_in[15];
    const float* bff2    = (const float*)d_in[16];
    const float* lnf_g   = (const float*)d_in[17];
    const float* lnf_b   = (const float*)d_in[18];
    const float* Wlm     = (const float*)d_in[19];
    const float* blm     = (const float*)d_in[20];
    float* outl = (float*)d_out;

    char* p = (char*)d_ws;
    auto alloc = [&](size_t bytes) -> char* {
        char* r = p; p += (bytes + 255) & ~(size_t)255; return r;
    };
    __bf16* wq_bf   = (__bf16*)alloc((size_t)L * C * C * 2);
    __bf16* wk_bf   = (__bf16*)alloc((size_t)L * C * C * 2);
    __bf16* wv_bf   = (__bf16*)alloc((size_t)L * C * C * 2);
    __bf16* wp_bf   = (__bf16*)alloc((size_t)L * C * C * 2);
    __bf16* wff1_bf = (__bf16*)alloc((size_t)L * C * FF * 2);
    __bf16* wff2_bf = (__bf16*)alloc((size_t)L * FF * C * 2);
    __bf16* wlm_bf  = (__bf16*)alloc((size_t)C * V * 2);
    float*  x       = (float*)alloc((size_t)BT * C * 4);
    __bf16* hb      = (__bf16*)alloc((size_t)BT * C * 2);
    __bf16* qb      = (__bf16*)alloc((size_t)BT * C * 2);
    __bf16* kb_     = (__bf16*)alloc((size_t)BT * C * 2);
    __bf16* vb      = (__bf16*)alloc((size_t)BT * C * 2);
    __bf16* ob      = (__bf16*)alloc((size_t)BT * C * 2);
    __bf16* ffb     = (__bf16*)alloc((size_t)BT * FF * 2);

    auto cvt = [&](const float* src, __bf16* dst, size_t n) {
        cvt_kernel<<<2048, 256, 0, stream>>>(src, dst, (int)n);
    };
    cvt(Wq, wq_bf, (size_t)L * C * C);
    cvt(Wk, wk_bf, (size_t)L * C * C);
    cvt(Wv, wv_bf, (size_t)L * C * C);
    cvt(Wproj, wp_bf, (size_t)L * C * C);
    cvt(Wff1, wff1_bf, (size_t)L * C * FF);
    cvt(Wff2, wff2_bf, (size_t)L * FF * C);
    cvt(Wlm, wlm_bf, (size_t)C * V);

    embed_kernel<<<BT * C / 256, 256, 0, stream>>>(inp_x, tok_emb, pos_emb, x, T, C);

    const float scale = 0.04419417382415922f;  // 512^-0.5 (reference scales by C)
    const dim3 g1(BT / 128, C / 128);          // 32 x 4
    const dim3 g2(BT / 128, FF / 128);         // 32 x 16
    const dim3 g3(BT / 128, V / 128);          // 32 x 250

    for (int l = 0; l < L; l++) {
        ln_kernel<<<BT, 128, 0, stream>>>(x, ln1_g + l * C, ln1_b + l * C, hb, C);
        gemm_bf16_kernel<false, false, false, true><<<g1, 256, 0, stream>>>(
            hb, wq_bf + (size_t)l * C * C, nullptr, nullptr, qb, BT, C, C);
        gemm_bf16_kernel<false, false, false, true><<<g1, 256, 0, stream>>>(
            hb, wk_bf + (size_t)l * C * C, nullptr, nullptr, kb_, BT, C, C);
        gemm_bf16_kernel<false, false, false, true><<<g1, 256, 0, stream>>>(
            hb, wv_bf + (size_t)l * C * C, nullptr, nullptr, vb, BT, C, C);
        attn_kernel<<<dim3(T / 16, H, Bb), 32, 0, stream>>>(qb, kb_, vb, ob, T, C, scale);
        gemm_bf16_kernel<true, false, true, false><<<g1, 256, 0, stream>>>(
            ob, wp_bf + (size_t)l * C * C, bproj + l * C, x, x, BT, C, C);
        ln_kernel<<<BT, 128, 0, stream>>>(x, ln2_g + l * C, ln2_b + l * C, hb, C);
        gemm_bf16_kernel<true, true, false, true><<<g2, 256, 0, stream>>>(
            hb, wff1_bf + (size_t)l * C * FF, bff1 + l * FF, nullptr, ffb, BT, FF, C);
        gemm_bf16_kernel<true, false, true, false><<<g1, 256, 0, stream>>>(
            ffb, wff2_bf + (size_t)l * FF * C, bff2 + l * C, x, x, BT, C, FF);
    }

    ln_kernel<<<BT, 128, 0, stream>>>(x, lnf_g, lnf_b, hb, C);
    gemm_bf16_kernel<true, false, false, false><<<g3, 256, 0, stream>>>(
        hb, wlm_bf, blm, nullptr, outl, BT, V, C);
}
